// GNNMambaLayer_31147102830925
// MI455X (gfx1250) — matrix operand order repaired and checked by reading, not compiled, not run on hardware
//
#include <hip/hip_runtime.h>
#include <hip/hip_bf16.h>
#include <math.h>

// ---------------------------------------------------------------------------
// GNN + bidirectional Mamba + MLP block for MI455X (gfx1250, wave32, WMMA).
// GEMMs: v_wmma_f32_16x16x32_f16. A tiles are DMA'd by the Tensor Data Mover
// into double-buffered LDS (issue tile kt+1, s_wait_tensorcnt 1 -> tile kt
// ready, overlap DMA with WMMA). B tiles ping-pong via vectorized LDS stage.
// ---------------------------------------------------------------------------

#define Nn   16384
#define Dd   256
#define Ee   262144
#define BGg  64
#define Ll   256
#define DIi  256
#define DSs  16
#define DTRr 16
#define DCc  4

typedef __attribute__((ext_vector_type(16))) _Float16 v16h;
typedef __attribute__((ext_vector_type(8)))  float    v8f;
typedef __attribute__((ext_vector_type(4)))  unsigned v4u;
typedef __attribute__((ext_vector_type(8)))  int      v8i;
typedef __attribute__((ext_vector_type(4)))  int      v4i;

enum { ACT_NONE = 0, ACT_GELU = 1, ACT_SILU = 2, ACT_SOFTPLUS = 3 };

__device__ __forceinline__ float siluf(float v) { return v / (1.0f + __expf(-v)); }

// ---------------------------------------------------------------------------
// Tensor Data Mover: DMA a 128(rows) x 32(cols) f32 tile (row stride =
// strideElems) from global into LDS at lds_off. Descriptor per ISA 08 S8.
// ---------------------------------------------------------------------------
__device__ __forceinline__ void tdm_load_a_tile(const float* gsrc, unsigned lds_off,
                                                int strideElems) {
    unsigned long long ga = (unsigned long long)(uintptr_t)gsrc;
    v4u g0;
    g0.x = 1u;                                             // count=1, user mode
    g0.y = lds_off;                                        // lds_addr
    g0.z = (unsigned)ga;                                   // global_addr[31:0]
    g0.w = (unsigned)((ga >> 32) & 0x1FFFFFFull) | (2u << 30); // [56:32] | type=2

    unsigned long long st = (unsigned long long)(unsigned)strideElems;
    // group1: data_size=4B(code 2), tensor_dim0=stride, tensor_dim1=1M,
    // tile_dim0=32, tile_dim1=128, tensor_dim0_stride=stride.
    unsigned long long q0 = (2ull << 16) | ((st & 0xFFFFull) << 48);
    unsigned long long q1 = ((st >> 16) & 0xFFFFull)
                          | (0x100000ull << 16)
                          | (32ull << 48);
    unsigned long long q2 = 128ull | ((st & 0xFFFFFFFFull) << 32);
    unsigned long long q3 = (st >> 32) & 0xFFFFull;

    v8i g1;
    g1[0] = (int)(unsigned)q0; g1[1] = (int)(q0 >> 32);
    g1[2] = (int)(unsigned)q1; g1[3] = (int)(q1 >> 32);
    g1[4] = (int)(unsigned)q2; g1[5] = (int)(q2 >> 32);
    g1[6] = (int)(unsigned)q3; g1[7] = (int)(q3 >> 32);
    v4i gz = {0, 0, 0, 0};
#if defined(__clang_major__) && (__clang_major__ >= 23)
    v8i gz8 = {0, 0, 0, 0, 0, 0, 0, 0};
    __builtin_amdgcn_tensor_load_to_lds(g0, g1, gz, gz, gz8, 0);
#else
    __builtin_amdgcn_tensor_load_to_lds(g0, g1, gz, gz, 0);
#endif
}

// ---------------------------------------------------------------------------
// Generic WMMA GEMM:  out[M, Nout] (+)= act( A[M,K](f32) @ W[Nout,K]^T + bias )
// Block = 256 threads = 8 waves; block tile = 128(M) x 16(N); each wave owns
// one 16x16 tile; K stepped by 32. TDM double-buffers the A tile in LDS; the
// kt+1 DMA overlaps the kt WMMA (s_wait_tensorcnt 1 before the barrier).
// ---------------------------------------------------------------------------
template <int ACTV, bool ACC, bool HAS_BIAS, bool KMUL32>
__global__ __launch_bounds__(256) void wmma_gemm(
    const float* __restrict__ A, int lda, int K,
    const float* __restrict__ W,          // [Nout, K] row-major
    const float* __restrict__ bias,       // [Nout] or nullptr
    float* __restrict__ out, int ldo)
{
    const int tid  = threadIdx.x;
    const int wave = tid >> 5;
    const int lane = tid & 31;
    const int mBase = blockIdx.x * 128;
    const int nBase = blockIdx.y * 16;

    __shared__ float    lA[2][128 * 32];  // 2 x 16 KB, f32 (TDM destinations)
    __shared__ _Float16 lB[2][16 * 32];   // 2 x  1 KB, f16

    v8f c = {};
    const int kTiles = (K + 31) / 32;

    if (KMUL32) {
        // ---- Prologue: stage tile 0 into buffer 0 -------------------------
        if (tid < 32)
            tdm_load_a_tile(A + (size_t)mBase * lda, (unsigned)(uintptr_t)&lA[0][0], lda);
        if (tid < 128) {
            const int n  = tid >> 3;
            const int k4 = (tid & 7) * 4;
            const float4 wv = *(const float4*)&W[(size_t)(nBase + n) * K + k4];
            _Float16* bp = &lB[0][n * 32 + k4];
            bp[0] = (_Float16)wv.x; bp[1] = (_Float16)wv.y;
            bp[2] = (_Float16)wv.z; bp[3] = (_Float16)wv.w;
        }

        for (int kt = 0; kt < kTiles; ++kt) {
            const int cur = kt & 1;
            // ---- Issue next tile's DMA into the other buffer --------------
            if (kt + 1 < kTiles) {
                const int k0n = (kt + 1) * 32;
                if (tid < 32)
                    tdm_load_a_tile(A + (size_t)mBase * lda + k0n,
                                    (unsigned)(uintptr_t)&lA[cur ^ 1][0], lda);
                if (tid < 128) {
                    const int n  = tid >> 3;
                    const int k4 = (tid & 7) * 4;
                    const float4 wv = *(const float4*)&W[(size_t)(nBase + n) * K + k0n + k4];
                    _Float16* bp = &lB[cur ^ 1][n * 32 + k4];
                    bp[0] = (_Float16)wv.x; bp[1] = (_Float16)wv.y;
                    bp[2] = (_Float16)wv.z; bp[3] = (_Float16)wv.w;
                }
                // In-order TENSORcnt: <=1 outstanding => tile kt has landed.
                if (tid < 32) __builtin_amdgcn_s_wait_tensorcnt(1);
            } else {
                if (tid < 32) __builtin_amdgcn_s_wait_tensorcnt(0);
            }
            __syncthreads();

            // ---- Fragments (CDNA5 16-bit A 16x32 / B 32x16 layouts) -------
            const int m  = lane & 15;
            const int hi = lane >> 4;
            const float* aRow = &lA[cur][(wave * 16 + m) * 32 + hi * 8];
            v16h af, bf;
            #pragma unroll
            for (int h = 0; h < 8; ++h) af[h]     = (_Float16)aRow[h];
            #pragma unroll
            for (int h = 0; h < 8; ++h) af[8 + h] = (_Float16)aRow[16 + h];
            #pragma unroll
            for (int h = 0; h < 16; ++h) bf[h] = lB[cur][m * 32 + hi * 8 + (h & 7) + (h >> 3) * 16];

            c = __builtin_amdgcn_wmma_f32_16x16x32_f16(
                    false, af, false, bf, (short)0, c, false, false);
            __syncthreads();
        }
    } else {
        // Guarded single-buffer fallback for K % 32 != 0 (dt proj, K=16).
        for (int kt = 0; kt < kTiles; ++kt) {
            const int k0 = kt * 32;
            for (int e = tid; e < 128 * 32; e += 256) {
                const int r = e >> 5, k = e & 31;
                lA[0][e] = (k0 + k < K) ? A[(size_t)(mBase + r) * lda + k0 + k] : 0.0f;
            }
            for (int e = tid; e < 16 * 32; e += 256) {
                const int n = e >> 5, k = e & 31;
                float v = (k0 + k < K) ? W[(size_t)(nBase + n) * K + k0 + k] : 0.0f;
                lB[0][e] = (_Float16)v;
            }
            __syncthreads();
            const int m  = lane & 15;
            const int hi = lane >> 4;
            const float* aRow = &lA[0][(wave * 16 + m) * 32 + hi * 8];
            v16h af, bf;
            #pragma unroll
            for (int h = 0; h < 8; ++h) af[h]     = (_Float16)aRow[h];
            #pragma unroll
            for (int h = 0; h < 8; ++h) af[8 + h] = (_Float16)aRow[16 + h];
            #pragma unroll
            for (int h = 0; h < 16; ++h) bf[h] = lB[0][m * 32 + hi * 8 + (h & 7) + (h >> 3) * 16];
            c = __builtin_amdgcn_wmma_f32_16x16x32_f16(
                    false, af, false, bf, (short)0, c, false, false);
            __syncthreads();
        }
    }

    // Epilogue. C/D layout: VGPR r -> M = r + 8*hi, N = lane&15.
    const int n  = lane & 15;
    const int hi = lane >> 4;
    #pragma unroll
    for (int r = 0; r < 8; ++r) {
        const int row = mBase + wave * 16 + r + hi * 8;
        const int col = nBase + n;
        float v = c[r];
        if (HAS_BIAS) v += bias[col];
        if (ACTV == ACT_GELU)      v = 0.5f * v * (1.0f + erff(v * 0.70710678118654752f));
        else if (ACTV == ACT_SILU) v = siluf(v);
        else if (ACTV == ACT_SOFTPLUS) v = (v > 20.0f) ? v : __logf(1.0f + __expf(v));
        float* p = out + (size_t)row * ldo + col;
        if (ACC) *p += v; else *p = v;
    }
}

// ---------------------------------------------------------------------------
// Small utility kernels
// ---------------------------------------------------------------------------
__global__ void set_f32(float* p, float v, int n) {
    int i = blockIdx.x * blockDim.x + threadIdx.x;
    if (i < n) p[i] = v;
}

__global__ void degree_count(const int* __restrict__ dst, float* __restrict__ deg, int ne) {
    int i = blockIdx.x * blockDim.x + threadIdx.x;
    if (i < ne) atomicAdd(&deg[dst[i]], 1.0f);
}

__global__ void rsqrt_inplace(float* p, int n) {
    int i = blockIdx.x * blockDim.x + threadIdx.x;
    if (i < n) p[i] = rsqrtf(p[i]);
}

// GCN edge aggregation: agg[dst] += xw[src] * dinv[src]*dinv[dst]
__global__ void gcn_scatter(const int* __restrict__ src, const int* __restrict__ dst,
                            const float* __restrict__ dinv, const float* __restrict__ xw,
                            float* __restrict__ agg) {
    int tid = blockIdx.x * blockDim.x + threadIdx.x;
    int e = tid >> 6, cch = tid & 63;
    if (e >= Ee) return;
    int s = src[e], t = dst[e];
    float norm = dinv[s] * dinv[t];
    const float4 xv = ((const float4*)(xw + (size_t)s * Dd))[cch];
    float* o = agg + (size_t)t * Dd + cch * 4;
    atomicAdd(o + 0, xv.x * norm);
    atomicAdd(o + 1, xv.y * norm);
    atomicAdd(o + 2, xv.z * norm);
    atomicAdd(o + 3, xv.w * norm);
}

// LayerNorm over D=256; one 256-thread block per row.
template <int MODE>
__global__ __launch_bounds__(256) void ln_kernel(
    const float* __restrict__ a, const float* __restrict__ b,
    const float* __restrict__ x, const float* __restrict__ dinv,
    const float* __restrict__ gbias,
    const float* __restrict__ g, const float* __restrict__ beta,
    float* __restrict__ out)
{
    const int row = blockIdx.x, d = threadIdx.x;
    const size_t i = (size_t)row * Dd + d;
    float v;
    if (MODE == 0) {
        float dv = dinv[row];
        v = a[i] + b[i] * dv * dv + gbias[d] + x[i];
    } else if (MODE == 1) {
        v = a[i] + x[i];
    } else {
        v = a[i] + b[i];
    }
    __shared__ float red[256];
    red[d] = v; __syncthreads();
    #pragma unroll
    for (int s = 128; s > 0; s >>= 1) { if (d < s) red[d] += red[d + s]; __syncthreads(); }
    const float mu = red[0] * (1.0f / Dd);
    __syncthreads();
    const float t = v - mu;
    red[d] = t * t; __syncthreads();
    #pragma unroll
    for (int s = 128; s > 0; s >>= 1) { if (d < s) red[d] += red[d + s]; __syncthreads(); }
    const float var = red[0] * (1.0f / Dd);
    out[i] = t * rsqrtf(var + 1e-5f) * g[d] + beta[d];
}

// Depthwise causal conv (DC=4) + SiLU; reverse=1 runs the time-flipped branch
// in place: xc_bwd[l] = silu(b + sum_j xi[l+j] * w[3-j]).
__global__ __launch_bounds__(256) void conv_silu(
    const float* __restrict__ xz, const float* __restrict__ cw,
    const float* __restrict__ cb, float* __restrict__ xc, int reverse)
{
    int idx = blockIdx.x * blockDim.x + threadIdx.x;
    if (idx >= Nn * DIi) return;
    const int d = idx & (DIi - 1);
    const int row = idx >> 8;
    const int l = row & (Ll - 1);
    float acc = cb[d];
    if (!reverse) {
        #pragma unroll
        for (int k = 0; k < DCc; ++k) {
            int ls = l - (DCc - 1) + k;
            if (ls >= 0) acc += xz[(size_t)(row - l + ls) * (2 * DIi) + d] * cw[d * DCc + k];
        }
    } else {
        #pragma unroll
        for (int j = 0; j < DCc; ++j) {
            int ls = l + j;
            if (ls < Ll) acc += xz[(size_t)(row - l + ls) * (2 * DIi) + d] * cw[d * DCc + (DCc - 1 - j)];
        }
    }
    xc[(size_t)row * DIi + d] = siluf(acc);
}

// Selective scan: block per batch, 256 threads = DI channels, 16 states in
// registers; fuses D-skip and silu(z) gate; reverse walks l backwards.
__global__ __launch_bounds__(256) void mamba_scan(
    const float* __restrict__ dt, const float* __restrict__ xc,
    const float* __restrict__ xz, const float* __restrict__ dbc,
    const float* __restrict__ A_log, const float* __restrict__ Dp,
    float* __restrict__ y, int reverse)
{
    const int b = blockIdx.x, d = threadIdx.x;
    float an[DSs], h[DSs];
    #pragma unroll
    for (int n = 0; n < DSs; ++n) { an[n] = -__expf(A_log[d * DSs + n]); h[n] = 0.0f; }
    const float Dpd = Dp[d];
    __shared__ float sB[DSs], sC[DSs];
    for (int step = 0; step < Ll; ++step) {
        const int l = reverse ? (Ll - 1 - step) : step;
        const int row = b * Ll + l;
        __syncthreads();
        if (d < DSs)          sB[d]       = dbc[(size_t)row * 48 + DTRr + d];
        else if (d < 2 * DSs) sC[d - DSs] = dbc[(size_t)row * 48 + DTRr + DSs + (d - DSs)];
        __syncthreads();
        const float dtv = dt[(size_t)row * DIi + d];
        const float xv  = xc[(size_t)row * DIi + d];
        const float zv  = xz[(size_t)row * (2 * DIi) + DIi + d];
        float acc = 0.0f;
        #pragma unroll
        for (int n = 0; n < DSs; ++n) {
            h[n] = __expf(dtv * an[n]) * h[n] + dtv * sB[n] * xv;
            acc += h[n] * sC[n];
        }
        y[(size_t)row * DIi + d] = (acc + xv * Dpd) * siluf(zv);
    }
}

__global__ void add_inplace(float* __restrict__ a, const float* __restrict__ b, int n) {
    int i = blockIdx.x * blockDim.x + threadIdx.x;
    if (i < n) a[i] += b[i];
}

// ---------------------------------------------------------------------------
// Host-side orchestration
// ---------------------------------------------------------------------------
extern "C" void kernel_launch(void* const* d_in, const int* in_sizes, int n_in,
                              void* d_out, int out_size, void* d_ws, size_t ws_size,
                              hipStream_t stream) {
    (void)in_sizes; (void)n_in; (void)out_size; (void)ws_size;
    const float* x        = (const float*)d_in[0];
    const int*   eidx     = (const int*)  d_in[1];
    const float* gcn_w    = (const float*)d_in[3];
    const float* gcn_b    = (const float*)d_in[4];
    const float* n1_g     = (const float*)d_in[5];
    const float* n1_b     = (const float*)d_in[6];
    const float* n2_g     = (const float*)d_in[7];
    const float* n2_b     = (const float*)d_in[8];
    const float* n3_g     = (const float*)d_in[9];
    const float* n3_b     = (const float*)d_in[10];
    const float* in_proj  = (const float*)d_in[11];
    const float* conv_w   = (const float*)d_in[12];
    const float* conv_b   = (const float*)d_in[13];
    const float* x_proj   = (const float*)d_in[14];
    const float* dt_w     = (const float*)d_in[15];
    const float* dt_b     = (const float*)d_in[16];
    const float* A_log    = (const float*)d_in[17];
    const float* Dp       = (const float*)d_in[18];
    const float* out_proj = (const float*)d_in[19];
    const float* mlp_w1   = (const float*)d_in[20];
    const float* mlp_b1   = (const float*)d_in[21];
    const float* mlp_w2   = (const float*)d_in[22];
    const float* mlp_b2   = (const float*)d_in[23];
    float* out = (float*)d_out;

    const int* e_src = eidx;
    const int* e_dst = eidx + Ee;

    char* ws = (char*)d_ws;
    const size_t MB = 1ull << 20;
    float* dinv = (float*)(ws);              // 64 KB (1 MB slot)
    float* S0   = (float*)(ws + 1 * MB);     // 32 MB : xz  -> mlp hidden
    float* S1   = (float*)(ws + 33 * MB);    // 16 MB : xw  -> xc -> mlp2
    float* S2   = (float*)(ws + 49 * MB);    // 16 MB : agg -> dt
    float* S3   = (float*)(ws + 65 * MB);    // 16 MB : h1  -> xs
    float* S4   = (float*)(ws + 81 * MB);    // 16 MB : mamba fwd+bwd sum
    float* S5   = (float*)(ws + 97 * MB);    // 16 MB : y   -> h2
    float* S6   = (float*)(ws + 113 * MB);   //  3 MB : dbc

    const dim3 blk(256);
    const dim3 gM16(Nn / 128, Dd / 16);
    const dim3 gM32(Nn / 128, 2 * Dd / 16);
    const dim3 gDbc(Nn / 128, 48 / 16);

    // ---- GCN branch -------------------------------------------------------
    set_f32<<<(Nn + 255) / 256, blk, 0, stream>>>(dinv, 1.0f, Nn);
    degree_count<<<(Ee + 255) / 256, blk, 0, stream>>>(e_dst, dinv, Ee);
    rsqrt_inplace<<<(Nn + 255) / 256, blk, 0, stream>>>(dinv, Nn);

    wmma_gemm<ACT_NONE, false, false, true><<<gM16, blk, 0, stream>>>(
        x, Dd, Dd, gcn_w, nullptr, S1, Dd);

    set_f32<<<(Nn * Dd + 255) / 256, blk, 0, stream>>>(S2, 0.0f, Nn * Dd);
    gcn_scatter<<<(Ee * 64 + 255) / 256, blk, 0, stream>>>(e_src, e_dst, dinv, S1, S2);

    ln_kernel<0><<<Nn, blk, 0, stream>>>(S2, S1, x, dinv, gcn_b, n1_g, n1_b, S3);

    // ---- Bidirectional Mamba ---------------------------------------------
    wmma_gemm<ACT_NONE, false, false, true><<<gM32, blk, 0, stream>>>(
        x, Dd, Dd, in_proj, nullptr, S0, 2 * DIi);

    set_f32<<<(Nn * Dd + 255) / 256, blk, 0, stream>>>(S4, 0.0f, Nn * Dd);

    for (int rev = 0; rev < 2; ++rev) {
        conv_silu<<<(Nn * DIi + 255) / 256, blk, 0, stream>>>(S0, conv_w, conv_b, S1, rev);
        wmma_gemm<ACT_NONE, false, false, true><<<gDbc, blk, 0, stream>>>(
            S1, DIi, DIi, x_proj, nullptr, S6, 48);
        wmma_gemm<ACT_SOFTPLUS, false, true, false><<<gM16, blk, 0, stream>>>(
            S6, 48, DTRr, dt_w, dt_b, S2, DIi);
        mamba_scan<<<BGg, blk, 0, stream>>>(S2, S1, S0, S6, A_log, Dp, S5, rev);
        wmma_gemm<ACT_NONE, true, false, true><<<gM16, blk, 0, stream>>>(
            S5, DIi, DIi, out_proj, nullptr, S4, Dd);
    }

    ln_kernel<1><<<Nn, blk, 0, stream>>>(S4, nullptr, x, nullptr, nullptr, n2_g, n2_b, S5);

    // ---- xs = h1 + h2, MLP, final LN -------------------------------------
    add_inplace<<<(Nn * Dd + 255) / 256, blk, 0, stream>>>(S3, S5, Nn * Dd);

    wmma_gemm<ACT_GELU, false, true, true><<<gM32, blk, 0, stream>>>(
        S3, Dd, Dd, mlp_w1, mlp_b1, S0, 2 * Dd);
    wmma_gemm<ACT_NONE, false, true, true><<<gM16, blk, 0, stream>>>(
        S0, 2 * Dd, 2 * Dd, mlp_w2, mlp_b2, S1, Dd);

    ln_kernel<2><<<Nn, blk, 0, stream>>>(S3, S1, nullptr, nullptr, nullptr, n3_g, n3_b, out);
}
